// OTETM_68728066671324
// MI455X (gfx1250) — compile-verified, matrix operand order checked by reading, not run on hardware
//
#include <hip/hip_runtime.h>
#include <hip/hip_bf16.h>

typedef __attribute__((ext_vector_type(16))) _Float16 v16h;
typedef __attribute__((ext_vector_type(8)))  float    v8f;
typedef __attribute__((ext_vector_type(4)))  int      v4i;

#define B_ 512
#define V_ 50000
#define T_ 100
#define H_ 64
#define ZD_ 5
#define D_ 100
#define G_ 20
#define L_ 8
#define SAMPLE_ 20
#define RSK_ 20
#define SINK_ITERS_ 50
#define KPAD_ 128
#define NCHUNK_ 64
#define TPC_ 49            // 49*64 = 3136 >= 3125 column tiles
#define NTILES_ (V_/16)    // 3125 (V divisible by 16)
#define PI_F 3.14159265358979f

// ---------------- deterministic hash RNG ----------------
__device__ __forceinline__ unsigned hash3(unsigned a, unsigned b, unsigned c) {
  unsigned h = a * 0x9E3779B1u ^ b * 0x85EBCA77u ^ c * 0xC2B2AE3Du;
  h ^= h >> 16; h *= 0x7FEB352Du; h ^= h >> 15; h *= 0x846CA68Bu; h ^= h >> 16;
  return h;
}
__device__ __forceinline__ float u01(unsigned h) {
  return ((float)(h >> 8) + 0.5f) * (1.0f / 16777216.0f);  // in (0,1)
}

// K index pattern for 16-bit WMMA A/B fragments (ISA 7.12.2):
// VGPR j holds K pair {kpat, kpat+1}; upper half-wave offset +8.
__device__ __forceinline__ int kpat(int j, int half) {
  return ((j & 3) << 1) + ((j >> 2) << 4) + (half << 3);
}

// ---------------- f16 conversion / transpose kernels ----------------
// whT[n][k] = W_hidden[k][n] as f16  (K-contiguous for B fragments)
__global__ void k_convwh(const float* __restrict__ W, _Float16* __restrict__ whT) {
  size_t i = (size_t)blockIdx.x * blockDim.x + threadIdx.x;
  if (i >= (size_t)H_ * V_) return;
  size_t n = i / V_, k = i % V_;
  whT[i] = (_Float16)W[k * H_ + n];
}
// twb[v][k] = topic_word[k][v] as f16, zero-padded to K=128
__global__ void k_convtw(const float* __restrict__ tw, _Float16* __restrict__ twb) {
  size_t i = (size_t)blockIdx.x * blockDim.x + threadIdx.x;
  if (i >= (size_t)V_ * KPAD_) return;
  size_t v = i / KPAD_, k = i % KPAD_;
  twb[i] = (k < T_) ? (_Float16)tw[k * V_ + v] : (_Float16)0.0f;
}

// ---------------- GEMM1: h = dropout(relu(x @ W_hidden + b)) ----------------
// grid 32 (16-row stripes), block 128 = 4 waves; wave w owns column tile w*16.
// x tile staged in LDS as f32 via async-to-LDS (16B per thread per chunk);
// converted to f16 during fragment build. Tail chunk (V%32==16) peeled.
__global__ void __launch_bounds__(128)
k_gemm1(const float* __restrict__ x, const _Float16* __restrict__ whT,
        const float* __restrict__ bh, float* __restrict__ h) {
  __shared__ float Asf[16 * 32];
  const int m0 = blockIdx.x * 16;
  const int wave = threadIdx.x >> 5;
  const int lane = threadIdx.x & 31;
  const int half = lane >> 4;
  const int r = lane & 15;
  const int n0 = wave * 16;
  // per-thread staging coords: 4 consecutive floats of the 16x32 tile
  const int e4 = threadIdx.x * 4;
  const int srow = e4 >> 5, skk = e4 & 31;
  const float* sgp = x + (size_t)(m0 + srow) * V_ + skk;
  const _Float16* bbase = whT + (size_t)(n0 + r) * V_;

  v8f c = {};
  const int KFULL = V_ / 32;  // 1562 full chunks; tail has 16 valid K
  for (int kc = 0; kc < KFULL; ++kc) {
    const int k0 = kc * 32;
    __syncthreads();  // previous LDS readers done
#if __has_builtin(__builtin_amdgcn_global_load_async_to_lds_b128)
    __builtin_amdgcn_global_load_async_to_lds_b128(
        (__attribute__((address_space(1))) v4i*)(sgp + k0),
        (__attribute__((address_space(3))) v4i*)&Asf[e4], 0, 0);
#if __has_builtin(__builtin_amdgcn_s_wait_asynccnt)
    __builtin_amdgcn_s_wait_asynccnt(0);
#else
    asm volatile("s_wait_asynccnt 0x0" ::: "memory");
#endif
#else
    {  // sync fallback: vectorized 16B load + 16B LDS store
      float4 vq = *(const float4*)(sgp + k0);
      *(float4*)&Asf[e4] = vq;
    }
#endif
    // prefetch x 8 chunks ahead and whT 1 chunk ahead into near caches
    __builtin_prefetch(sgp + k0 + 8 * 32, 0, 3);
    __builtin_prefetch(bbase + k0 + 32, 0, 3);
    __syncthreads();
    v16h a, b;
    const _Float16* bp = bbase + k0;
    for (int j = 0; j < 8; ++j) {
      int k = kpat(j, half);
      a[2 * j]     = (_Float16)Asf[r * 32 + k];
      a[2 * j + 1] = (_Float16)Asf[r * 32 + k + 1];
      b[2 * j]     = bp[k];
      b[2 * j + 1] = bp[k + 1];
    }
    c = __builtin_amdgcn_wmma_f32_16x16x32_f16(false, a, false, b, (short)0, c, false, false);
  }
  {  // tail chunk: K columns 49984..49999 valid (16 of 32)
    const int k0 = KFULL * 32;
    __syncthreads();
    for (int q = 0; q < 4; ++q) {
      int kk2 = skk + q;
      Asf[e4 + q] = (kk2 < 16) ? sgp[k0 + kk2] : 0.0f;
    }
    __syncthreads();
    v16h a, b;
    const _Float16* bp = bbase + k0;
    for (int j = 0; j < 8; ++j) {
      int k = kpat(j, half);
      bool ok = (k < 16);  // pairs fall entirely in [0,16) or [16,32)
      a[2 * j]     = (_Float16)Asf[r * 32 + k];
      a[2 * j + 1] = (_Float16)Asf[r * 32 + k + 1];
      b[2 * j]     = ok ? bp[k]     : (_Float16)0.0f;
      b[2 * j + 1] = ok ? bp[k + 1] : (_Float16)0.0f;
    }
    c = __builtin_amdgcn_wmma_f32_16x16x32_f16(false, a, false, b, (short)0, c, false, false);
  }
  for (int i = 0; i < 8; ++i) {  // C layout: VGPR i -> M=i+half*8, N=r
    int row = m0 + i + half * 8, col = n0 + r;
    float v = fmaxf(c[i] + bh[col], 0.0f);
    v = (hash3((unsigned)row, (unsigned)col, 0x1234567u) & 1u) ? 2.0f * v : 0.0f;  // dropout p=.5
    h[row * H_ + col] = v;
  }
}

// ---------------- small chain: z_mean, kappa, KL, vMF sample, theta ----------------
__global__ void __launch_bounds__(512)
k_chain(const float* __restrict__ h, const float* __restrict__ Wm, const float* __restrict__ bm,
        const float* __restrict__ Wv, const float* __restrict__ bv,
        const float* __restrict__ Wz, const float* __restrict__ bz,
        _Float16* __restrict__ th16, float* __restrict__ scal) {
  __shared__ float red[512];
  const int b = threadIdx.x;
  float hv[H_];
  for (int j = 0; j < H_; ++j) hv[j] = h[b * H_ + j];
  // z_mean (normalized)
  float zm[ZD_]; float nn = 0.0f;
  for (int d = 0; d < ZD_; ++d) {
    float s = bm[d];
    for (int j = 0; j < H_; ++j) s += hv[j] * Wm[j * ZD_ + d];
    zm[d] = s; nn += s * s;
  }
  nn = rsqrtf(nn + 1e-30f);
  for (int d = 0; d < ZD_; ++d) zm[d] *= nn;
  // kappa = softplus(h@Wv + bv) + 1
  float kv = bv[0];
  for (int j = 0; j < H_; ++j) kv += hv[j] * Wv[j];
  float kap = ((kv > 20.0f) ? kv : log1pf(__expf(kv))) + 1.0f;
  // vMF KL (m = 5, half-integer Bessel closed form)
  {
    float k = kap;
    float t = tanhf(k);
    float ratio = (t * (1.0f + 3.0f / (k * k)) - 3.0f / k) / (1.0f - t / k);
    float e2k = __expf(-2.0f * k);
    float log_i32 = 0.5f * (logf(2.0f) - logf(PI_F) - logf(k)) + k - logf(2.0f)
                  + logf((1.0f + e2k) - (1.0f - e2k) / k);
    float log_c = 1.5f * logf(k) - 2.5f * logf(2.0f * PI_F) - log_i32;
    const float log_area = 3.2702897f;  // log 2 + 2.5 log pi - lgamma(2.5)
    red[b] = k * ratio + log_c + log_area;
  }
  __syncthreads();
  for (int s = 256; s; s >>= 1) {
    if (b < s) red[b] += red[b + s];
    __syncthreads();
  }
  if (b == 0) scal[0] = red[0] * (1.0f / B_);  // kld mean
  // Wood's rejection sampler for w (Beta(2,2) = median of 3 uniforms)
  float sq = sqrtf(4.0f * kap * kap + 16.0f);
  float bb = (-2.0f * kap + sq) * 0.25f;
  float aa = (4.0f + 2.0f * kap + sq) * 0.25f;
  float dd = 4.0f * aa * bb / (1.0f + bb) - 4.0f * logf(4.0f);
  float w = 0.0f; int done = 0;
  for (int rr = 0; rr < RSK_; ++rr) {
    float u1 = u01(hash3(b, rr, 0xA1u)), u2 = u01(hash3(b, rr, 0xA2u));
    float u3 = u01(hash3(b, rr, 0xA3u)), uu = u01(hash3(b, rr, 0xA4u));
    float eps = fmaxf(fminf(u1, u2), fminf(fmaxf(u1, u2), u3));
    float den = 1.0f - (1.0f - bb) * eps;
    float wr = (1.0f - (1.0f + bb) * eps) / den;
    float tc = 2.0f * aa * bb / den;
    int acc = (4.0f * logf(tc) - tc + dd) >= logf(uu);
    if (rr == 0) w = wr;                 // argmax(all false) -> round 0
    if (!done && acc) { w = wr; done = 1; }
  }
  // tangent direction (Box-Muller normals, orthogonalize, normalize)
  float vv[ZD_];
  for (int d = 0; d < ZD_; d += 2) {
    float u1 = u01(hash3(b, d, 0xB1u)), u2 = u01(hash3(b, d, 0xB2u));
    float rd = sqrtf(-2.0f * logf(u1));
    float sn, cn; __sincosf(6.2831853f * u2, &sn, &cn);
    vv[d] = rd * cn;
    if (d + 1 < ZD_) vv[d + 1] = rd * sn;
  }
  float dp = 0.0f;
  for (int d = 0; d < ZD_; ++d) dp += vv[d] * zm[d];
  float vn = 0.0f;
  for (int d = 0; d < ZD_; ++d) { vv[d] -= dp * zm[d]; vn += vv[d] * vv[d]; }
  vn = rsqrtf(vn + 1e-30f);
  float wc = sqrtf(fmaxf(1.0f - w * w, 0.0f));
  float z[ZD_];
  for (int d = 0; d < ZD_; ++d) z[d] = w * zm[d] + wc * vv[d] * vn;
  // theta = softmax(10z @ W_h2z + b), stored f16, K padded to 128
  float mxl = -3.0e38f;
  for (int t = 0; t < T_; ++t) {
    float s = bz[t];
    for (int d = 0; d < ZD_; ++d) s += 10.0f * z[d] * Wz[d * T_ + t];
    mxl = fmaxf(mxl, s);
  }
  float se = 0.0f;
  for (int t = 0; t < T_; ++t) {
    float s = bz[t];
    for (int d = 0; d < ZD_; ++d) s += 10.0f * z[d] * Wz[d * T_ + t];
    se += __expf(s - mxl);
  }
  float inv = 1.0f / se;
  for (int t = 0; t < KPAD_; ++t) {
    float s = 0.0f;
    if (t < T_) {
      float l = bz[t];
      for (int d = 0; d < ZD_; ++d) l += 10.0f * z[d] * Wz[d * T_ + t];
      s = __expf(l - mxl) * inv;
    }
    th16[(size_t)b * KPAD_ + t] = (_Float16)s;
  }
}

// ---------------- GEMM2 fused: logits tiles + online logsumexp + x-dot ----------------
// grid (32 row tiles, 64 V-chunks); block = 1 wave.
__global__ void __launch_bounds__(32)
k_gemm2(const float* __restrict__ x, const _Float16* __restrict__ th16,
        const _Float16* __restrict__ twb, float* __restrict__ part) {
  __shared__ float ct[256];
  const int m0 = blockIdx.x * 16;
  const int chunk = blockIdx.y;
  const int lane = threadIdx.x;
  const int half = lane >> 4, r = lane & 15;
  v16h a0, a1, a2, a3;
  {  // A fragments (theta rows, K = 0..127), loaded once
    const _Float16* tp = th16 + (size_t)(m0 + r) * KPAD_;
    for (int j = 0; j < 8; ++j) {
      int k = kpat(j, half);
      a0[2*j] = tp[k];      a0[2*j+1] = tp[k+1];
      a1[2*j] = tp[32+k];   a1[2*j+1] = tp[32+k+1];
      a2[2*j] = tp[64+k];   a2[2*j+1] = tp[64+k+1];
      a3[2*j] = tp[96+k];   a3[2*j+1] = tp[96+k+1];
    }
  }
  float mx = -3.0e38f, ss = 0.0f, dot = 0.0f, sx = 0.0f;
  for (int ti = 0; ti < TPC_; ++ti) {
    int vt = chunk * TPC_ + ti;
    if (vt >= NTILES_) break;
    int v0 = vt * 16;
    const _Float16* bp = twb + (size_t)(v0 + r) * KPAD_;
    __builtin_prefetch(bp + 16 * KPAD_, 0, 3);  // next tile's B row for this lane
    v16h b0, b1, b2, b3;
    for (int j = 0; j < 8; ++j) {
      int k = kpat(j, half);
      b0[2*j] = bp[k];      b0[2*j+1] = bp[k+1];
      b1[2*j] = bp[32+k];   b1[2*j+1] = bp[32+k+1];
      b2[2*j] = bp[64+k];   b2[2*j+1] = bp[64+k+1];
      b3[2*j] = bp[96+k];   b3[2*j+1] = bp[96+k+1];
    }
    v8f c = {};
    c = __builtin_amdgcn_wmma_f32_16x16x32_f16(false, a0, false, b0, (short)0, c, false, false);
    c = __builtin_amdgcn_wmma_f32_16x16x32_f16(false, a1, false, b1, (short)0, c, false, false);
    c = __builtin_amdgcn_wmma_f32_16x16x32_f16(false, a2, false, b2, (short)0, c, false, false);
    c = __builtin_amdgcn_wmma_f32_16x16x32_f16(false, a3, false, b3, (short)0, c, false, false);
    __syncthreads();
    for (int i = 0; i < 8; ++i) ct[(i + half * 8) * 16 + r] = c[i];
    __syncthreads();
    if (lane < 16) {  // one thread per logits row: online lse + x dot
      const float* xp = x + (size_t)(m0 + lane) * V_ + v0;
      __builtin_prefetch(xp + 16, 0, 3);
      for (int n = 0; n < 16; ++n) {
        float l = ct[lane * 16 + n];
        float xv = xp[n];
        float nm = fmaxf(mx, l);
        ss = ss * __expf(mx - nm) + __expf(l - nm);
        mx = nm;
        dot += xv * l;
        sx += xv;
      }
    }
  }
  if (lane < 16) {
    float* p = part + (((size_t)blockIdx.x * NCHUNK_ + chunk) * 16 + lane) * 4;
    p[0] = mx; p[1] = ss; p[2] = dot; p[3] = sx;
  }
}

// merge chunk partials -> rec_loss[b] = sx*lse - dot
__global__ void k_merge(const float* __restrict__ part, float* __restrict__ rec) {
  int b = blockIdx.x * blockDim.x + threadIdx.x;
  if (b >= B_) return;
  int rt = b >> 4, r = b & 15;
  float m = -3.0e38f, s = 0.0f, dot = 0.0f, sx = 0.0f;
  for (int c = 0; c < NCHUNK_; ++c) {
    const float* p = part + (((size_t)rt * NCHUNK_ + c) * 16 + r) * 4;
    float mi = p[0], si = p[1];
    float nm = fmaxf(m, mi);
    s = s * __expf(m - nm) + si * __expf(mi - nm);
    m = nm;
    dot += p[2]; sx += p[3];
  }
  rec[b] = sx * (m + logf(s)) - dot;
}

// ---------------- OT prior pieces ----------------
__global__ void __launch_bounds__(256)
k_rowlse(const float* __restrict__ tw, float* __restrict__ lse) {
  __shared__ float red[256];
  const int t = blockIdx.x;
  const float* row = tw + (size_t)t * V_;
  float m = -3.0e38f;
  for (int v = threadIdx.x; v < V_; v += 256) m = fmaxf(m, row[v]);
  red[threadIdx.x] = m; __syncthreads();
  for (int s = 128; s; s >>= 1) { if (threadIdx.x < s) red[threadIdx.x] = fmaxf(red[threadIdx.x], red[threadIdx.x + s]); __syncthreads(); }
  m = red[0]; __syncthreads();
  float su = 0.0f;
  for (int v = threadIdx.x; v < V_; v += 256) su += __expf(row[v] - m);
  red[threadIdx.x] = su; __syncthreads();
  for (int s = 128; s; s >>= 1) { if (threadIdx.x < s) red[threadIdx.x] += red[threadIdx.x + s]; __syncthreads(); }
  if (threadIdx.x == 0) lse[t] = m + logf(red[0]);
}

__global__ void k_gmcost(const float* __restrict__ tw, const float* __restrict__ lse,
                         const int* __restrict__ index, float* __restrict__ cost,
                         float* __restrict__ val) {
  for (int id = threadIdx.x; id < T_ * G_; id += blockDim.x) {
    int t = id / G_, g = id % G_;
    float s = 0.0f;
    for (int l = 0; l < L_; ++l) {
      int vi = index[g * L_ + l];
      s += -logf(__expf(tw[(size_t)t * V_ + vi] - lse[t]) + 1e-12f);
    }
    cost[g * T_ + t] = s * (1.0f / L_);   // cost = gm.T
  }
  for (int t = threadIdx.x; t < T_; t += blockDim.x) {
    float vmax = -3.0e38f;
    for (int i = 0; i < G_ * L_; ++i) {
      int vi = index[i];
      vmax = fmaxf(vmax, logf(__expf(tw[(size_t)t * V_ + vi] - lse[t]) + 1e-12f));
    }
    val[t] = vmax;
  }
}

__global__ void __launch_bounds__(128)
k_sink(const float* __restrict__ cost, float* __restrict__ scal) {
  __shared__ float Mc[G_ * T_];
  __shared__ float fv[G_], gv[T_];
  __shared__ float red[128];
  for (int i = threadIdx.x; i < G_ * T_; i += 128) Mc[i] = cost[i];
  if (threadIdx.x < G_) fv[threadIdx.x] = 0.0f;
  if (threadIdx.x < T_) gv[threadIdx.x] = 0.0f;
  __syncthreads();
  const float lam = 20.0f, reg = 0.05f;
  for (int it = 0; it < SINK_ITERS_; ++it) {
    if (threadIdx.x < G_) {
      int g = threadIdx.x;
      float m = -3.0e38f;
      for (int j = 0; j < T_; ++j) m = fmaxf(m, (gv[j] - Mc[g * T_ + j]) * lam);
      float s = 0.0f;
      for (int j = 0; j < T_; ++j) s += __expf((gv[j] - Mc[g * T_ + j]) * lam - m);
      fv[g] = -reg * (m + logf(s));
    }
    __syncthreads();
    if (threadIdx.x < T_) {
      int j = threadIdx.x;
      float m = -3.0e38f;
      for (int g = 0; g < G_; ++g) m = fmaxf(m, (fv[g] - Mc[g * T_ + j]) * lam);
      float s = 0.0f;
      for (int g = 0; g < G_; ++g) s += __expf((fv[g] - Mc[g * T_ + j]) * lam - m);
      gv[j] = -reg * (m + logf(s));
    }
    __syncthreads();
  }
  float acc = 0.0f;
  for (int i = threadIdx.x; i < G_ * T_; i += 128) {
    int g = i / T_, j = i % T_;
    acc += __expf((fv[g] + gv[j] - Mc[i]) * lam) * Mc[i];
  }
  red[threadIdx.x] = acc; __syncthreads();
  for (int s = 64; s; s >>= 1) { if (threadIdx.x < s) red[threadIdx.x] += red[threadIdx.x + s]; __syncthreads(); }
  if (threadIdx.x == 0) scal[1] = red[0];
}

// per-group: argmax topic, top-20 over its row, similarity-gated mean
__global__ void __launch_bounds__(256)
k_topk(const float* __restrict__ tw, const float* __restrict__ lse,
       const float* __restrict__ cost, const float* __restrict__ val,
       const int* __restrict__ index, const float* __restrict__ wemb,
       float* __restrict__ nsl) {
  __shared__ int sg_s;
  __shared__ float cv[256 * SAMPLE_];
  __shared__ int   ci[256 * SAMPLE_];
  __shared__ float bestv[SAMPLE_];
  __shared__ int   besti[SAMPLE_];
  __shared__ float sred[SAMPLE_], cred[SAMPLE_];
  const int g = blockIdx.x;
  if (threadIdx.x == 0) {
    float best = -3.0e38f; int bi = 0;
    for (int t = 0; t < T_; ++t) {
      float sc = -cost[g * T_ + t] - val[t];
      if (sc > best) { best = sc; bi = t; }
    }
    sg_s = bi;
  }
  __syncthreads();
  const int t = sg_s;
  const float* row = tw + (size_t)t * V_;
  float lv[SAMPLE_]; int li[SAMPLE_];
  for (int s = 0; s < SAMPLE_; ++s) { lv[s] = -3.0e38f; li[s] = -1; }
  for (int v = threadIdx.x; v < V_; v += 256) {  // logs monotone in tw -> top-k of raw row
    float xv = row[v];
    if (xv > lv[SAMPLE_ - 1]) {
      int p = SAMPLE_ - 1;
      while (p > 0 && lv[p - 1] < xv) { lv[p] = lv[p - 1]; li[p] = li[p - 1]; --p; }
      lv[p] = xv; li[p] = v;
    }
  }
  for (int s = 0; s < SAMPLE_; ++s) { cv[threadIdx.x * SAMPLE_ + s] = lv[s]; ci[threadIdx.x * SAMPLE_ + s] = li[s]; }
  __syncthreads();
  if (threadIdx.x == 0) {
    for (int s = 0; s < SAMPLE_; ++s) {
      float best = -3.0e38f; int bi = 0;
      for (int i = 0; i < 256 * SAMPLE_; ++i)
        if (cv[i] > best) { best = cv[i]; bi = i; }
      bestv[s] = best; besti[s] = ci[bi]; cv[bi] = -3.0e38f;
    }
  }
  __syncthreads();
  if (threadIdx.x < SAMPLE_) {
    int s = threadIdx.x;
    float nsv = logf(__expf(bestv[s] - lse[t]) + 1e-12f);
    int nv = besti[s];
    float smax = -3.0e38f;
    for (int l = 0; l < L_; ++l) {
      int ev = index[g * L_ + l];
      float d = 0.0f;
      for (int dd = 0; dd < D_; ++dd) d += wemb[(size_t)nv * D_ + dd] * wemb[(size_t)ev * D_ + dd];
      smax = fmaxf(smax, d);
    }
    float prob = fminf(fmaxf(1.0f - smax, 0.0f), 1.0f);
    float samp = (u01(hash3(g, s, 0xC5u)) < prob) ? 1.0f : 0.0f;
    sred[s] = nsv * samp;
    cred[s] = samp;
  }
  __syncthreads();
  if (threadIdx.x == 0) {
    float cnt = 0.0f, ws = 0.0f;
    for (int s = 0; s < SAMPLE_; ++s) { cnt += cred[s]; ws += sred[s]; }
    nsl[g] = (cnt > 0.0f) ? ws / fmaxf(cnt, 1.0f) : 0.0f;
  }
}

// ---------------- diversity penalty ----------------
__global__ void __launch_bounds__(256)
k_div(const float* __restrict__ temb, float* __restrict__ scal) {
  __shared__ float nt[T_ * D_];
  __shared__ float red[256], red2[256];
  for (int i = threadIdx.x; i < T_ * D_; i += 256) nt[i] = temb[i];
  __syncthreads();
  if (threadIdx.x < T_) {
    int r = threadIdx.x;
    float n = 0.0f;
    for (int d = 0; d < D_; ++d) n += nt[r * D_ + d] * nt[r * D_ + d];
    n = 1.0f / (sqrtf(n) + 1e-12f);
    for (int d = 0; d < D_; ++d) nt[r * D_ + d] *= n;
  }
  __syncthreads();
  float s1 = 0.0f, s2 = 0.0f;
  for (int e = threadIdx.x; e < T_ * T_; e += 256) {
    int i = e / T_, j = e % T_;
    float d = 0.0f;
    for (int dd = 0; dd < D_; ++dd) d += nt[i * D_ + dd] * nt[j * D_ + dd];
    d = fabsf(d);
    s1 += d; s2 += d * d;
  }
  red[threadIdx.x] = s1; red2[threadIdx.x] = s2; __syncthreads();
  for (int s = 128; s; s >>= 1) {
    if (threadIdx.x < s) { red[threadIdx.x] += red[threadIdx.x + s]; red2[threadIdx.x] += red2[threadIdx.x + s]; }
    __syncthreads();
  }
  if (threadIdx.x == 0) {
    float cm = red[0] * (1.0f / (T_ * T_));
    float cvr = red2[0] * (1.0f / (T_ * T_)) - cm * cm;
    scal[2] = cm - cvr;
  }
}

// ---------------- final combine ----------------
__global__ void k_final(const float* __restrict__ rec, const float* __restrict__ scal,
                        const float* __restrict__ nsl, const int* __restrict__ epoch,
                        float* __restrict__ out) {
  int b = blockIdx.x * blockDim.x + threadIdx.x;
  if (b >= B_) return;
  int ep = epoch[0];
  float pp = 0.0f;
  if (ep >= 10) {
    pp = scal[1];            // BETA * sinkhorn, BETA = 1
    if (ep > 20) {
      float s = 0.0f;
      for (int g = 0; g < G_; ++g) s += nsl[g];
      pp += s;               // GAMMA = 1
    }
  }
  out[b] = rec[b] + scal[0] + pp + scal[2] * 0.1f;
}

// ---------------- launcher ----------------
extern "C" void kernel_launch(void* const* d_in, const int* in_sizes, int n_in,
                              void* d_out, int out_size, void* d_ws, size_t ws_size,
                              hipStream_t stream) {
  const float* x   = (const float*)d_in[0];
  const float* Wh  = (const float*)d_in[1];
  const float* bh  = (const float*)d_in[2];
  const float* Wm  = (const float*)d_in[3];
  const float* bm  = (const float*)d_in[4];
  const float* Wv  = (const float*)d_in[5];
  const float* bv  = (const float*)d_in[6];
  const float* Wz  = (const float*)d_in[7];
  const float* bz  = (const float*)d_in[8];
  const float* tw  = (const float*)d_in[9];
  const float* te  = (const float*)d_in[10];
  const float* we  = (const float*)d_in[11];
  const int*   idx = (const int*)d_in[12];
  const int*   ep  = (const int*)d_in[13];
  float* out = (float*)d_out;

  char* base = (char*)d_ws;
  const size_t OFF_TWB  = 0;                                    // f16 [V][128]  12.8 MB
  const size_t OFF_WHT  = OFF_TWB + (size_t)V_ * KPAD_ * 2;     // f16 [H][V]    6.4 MB
  const size_t OFF_TH16 = OFF_WHT + (size_t)H_ * V_ * 2;        // f16 [B][128]
  const size_t OFF_H    = OFF_TH16 + (size_t)B_ * KPAD_ * 2;    // f32 [B][H]
  const size_t OFF_REC  = OFF_H + (size_t)B_ * H_ * 4;
  const size_t OFF_PART = OFF_REC + 4096;                       // f32 [32][64][16][4]
  const size_t OFF_LSE  = OFF_PART + (size_t)32 * NCHUNK_ * 16 * 4 * 4;
  const size_t OFF_COST = OFF_LSE + 1024;                       // f32 [G][T]
  const size_t OFF_VAL  = OFF_COST + 8192;
  const size_t OFF_SCAL = OFF_VAL + 1024;                       // [0]=kld_mean [1]=sink [2]=dpen
  const size_t OFF_NSL  = OFF_SCAL + 256;

  _Float16* twb  = (_Float16*)(base + OFF_TWB);
  _Float16* whT  = (_Float16*)(base + OFF_WHT);
  _Float16* th16 = (_Float16*)(base + OFF_TH16);
  float* hbuf = (float*)(base + OFF_H);
  float* rec  = (float*)(base + OFF_REC);
  float* part = (float*)(base + OFF_PART);
  float* lse  = (float*)(base + OFF_LSE);
  float* cost = (float*)(base + OFF_COST);
  float* val  = (float*)(base + OFF_VAL);
  float* scal = (float*)(base + OFF_SCAL);
  float* nsl  = (float*)(base + OFF_NSL);

  (void)in_sizes; (void)n_in; (void)out_size; (void)ws_size;

  { size_t n = (size_t)H_ * V_;      k_convwh<<<(unsigned)((n + 255) / 256), 256, 0, stream>>>(Wh, whT); }
  { size_t n = (size_t)V_ * KPAD_;   k_convtw<<<(unsigned)((n + 255) / 256), 256, 0, stream>>>(tw, twb); }
  k_gemm1<<<32, 128, 0, stream>>>(x, whT, bh, hbuf);
  k_chain<<<1, 512, 0, stream>>>(hbuf, Wm, bm, Wv, bv, Wz, bz, th16, scal);
  k_gemm2<<<dim3(32, NCHUNK_), 32, 0, stream>>>(x, th16, twb, part);
  k_merge<<<2, 256, 0, stream>>>(part, rec);
  k_rowlse<<<T_, 256, 0, stream>>>(tw, lse);
  k_gmcost<<<1, 256, 0, stream>>>(tw, lse, idx, cost, val);
  k_sink<<<1, 128, 0, stream>>>(cost, scal);
  k_topk<<<G_, 256, 0, stream>>>(tw, lse, cost, val, idx, we, nsl);
  k_div<<<1, 256, 0, stream>>>(te, scal);
  k_final<<<2, 256, 0, stream>>>(rec, scal, nsl, ep, out);
}
